// PCNN_attention_69114613728837
// MI455X (gfx1250) — matrix operand order repaired and checked by reading, not compiled, not run on hardware
//
#include <hip/hip_runtime.h>
#include <hip/hip_bf16.h>
#include <math.h>

typedef __attribute__((ext_vector_type(16))) _Float16 v16h;
typedef __attribute__((ext_vector_type(8)))  _Float16 v8h;
typedef __attribute__((ext_vector_type(8)))  float    v8f;

#define OUT_C 230
#define N_REL 53
#define KT    6     // K' tiles: 192 / 32
#define NT    15    // N  tiles: 240 / 16
#define MT    4     // M  tiles:  64 / 16
#define BFRAG_ELEMS (KT * NT * 32 * 16)   // 46080 halfs = 92160 B

// Fast tanh: hardware V_TANH_F32 on gfx1250 if the builtin exists, else a
// branchless exp-based identity (clamped so e never overflows -> no NaN).
static __device__ __forceinline__ float fast_tanh(float x) {
#if __has_builtin(__builtin_amdgcn_tanhf)
  return __builtin_amdgcn_tanhf(x);
#else
  x = fminf(15.f, fmaxf(-15.f, x));
  float e = __expf(2.f * x);            // v_exp_f32 path, no slow branches
  return (e - 1.f) / (e + 1.f);
#endif
}

// ---------------------------------------------------------------------------
// Kernel 0: pre-swizzle conv weights into WMMA B-fragment layout (f32 -> f16).
// Bfrag[kt][nt][lane][e] = B[k' = kt*32 + 8*(lane>=16) + (e&7) + 16*(e>>3)][n = nt*16 + lane&15]
// where k' -> (kk = k'/64, f = k'%64), zero for f >= 60 or n >= 230.
// ---------------------------------------------------------------------------
__global__ void pcnn_pack_w(const float* __restrict__ conv_w,
                            _Float16* __restrict__ Bfrag) {
  int idx = blockIdx.x * 256 + threadIdx.x;
  if (idx >= BFRAG_ELEMS) return;
  int e  = idx & 15;
  int l  = (idx >> 4) & 31;
  int nt = (idx >> 9) % NT;
  int kt = (idx >> 9) / NT;
  int n  = nt * 16 + (l & 15);
  int kp = kt * 32 + ((l >> 4) << 3) + (e & 7) + ((e >> 3) << 4);   // k' in [0,192)
  int kk = kp >> 6;                         // conv tap 0..2
  int f  = kp & 63;                         // feature 0..63 (60..63 padded)
  float v = 0.f;
  if (n < OUT_C && f < 60) v = conv_w[(n * 3 + kk) * 60 + f];
  Bfrag[idx] = (_Float16)v;
}

// ---------------------------------------------------------------------------
// Kernel 1: per-sentence  embed-gather -> LDS(f16, stride 64) -> WMMA GEMM
//           -> tanh -> max-pool over positions.  One block per sentence.
// ---------------------------------------------------------------------------
__global__ __launch_bounds__(256)
void pcnn_conv(const int* __restrict__ tokens, const int* __restrict__ pos1,
               const int* __restrict__ pos2,  const int* __restrict__ sent_lens,
               const float* __restrict__ word_embed,
               const float* __restrict__ pos_embed,
               const float* __restrict__ conv_b,
               const _Float16* __restrict__ Bfrag,
               float* __restrict__ feats) {
  __shared__ _Float16 featflat[4224];       // 66 rows * 64, max A read = 63*64+191
  const int sid = blockIdx.x;               // b*16 + s
  const int tid = threadIdx.x;
  const int p1 = pos1[sid], p2 = pos2[sid], slen = sent_lens[sid];

  // Build zero-padded f16 feature map: feat[t][0:50]=word, [50:55]=pos1, [55:60]=pos2.
  for (int idx = tid; idx < 4224; idx += 256) {
    int t = idx >> 6, f = idx & 63;
    float v = 0.f;
    if (t < 60 && t < slen && f < 60) {
      if (f < 50) {
        v = word_embed[tokens[sid * 60 + t] * 50 + f];
      } else if (f < 55) {
        int d = t - p1; d = (d < -60 ? -60 : (d > 60 ? 60 : d)) + 60;
        v = pos_embed[d * 5 + (f - 50)];
      } else {
        int d = t - p2; d = (d < -60 ? -60 : (d > 60 ? 60 : d)) + 60;
        v = pos_embed[d * 5 + (f - 55)];
      }
    }
    featflat[idx] = (_Float16)v;
  }
  __syncthreads();

  const int wave  = tid >> 5;
  const int l     = tid & 31;
  const int lhalf = l >> 4;                  // 0 or 1

  // Each wave owns N-tiles nt = wave, wave+8 (uniform control flow -> EXEC all 1s).
  for (int nt = wave; nt < NT; nt += 8) {
    const int n = nt * 16 + (l & 15);
    const float bias = (n < OUT_C) ? conv_b[n] : 0.f;
    __builtin_prefetch(Bfrag + ((nt * 32) << 4), 0, 0);   // global_prefetch_b8
    float mx = -3.4e38f;

    for (int mt = 0; mt < MT; ++mt) {
      v8f acc = {};
      #pragma unroll
      for (int kt = 0; kt < KT; ++kt) {
        // A fragment: two aligned 16B LDS chunks per lane (ISA 16-bit A layout).
        const _Float16* ap =
            featflat + ((mt * 16 + (l & 15)) * 64 + kt * 32 + lhalf * 8);
        union { v16h v; v8h h[2]; } a;
        a.h[0] = *(const v8h*)ap;            // e=0..7  -> K = klo .. klo+7
        a.h[1] = *(const v8h*)(ap + 16);     // e=8..15 -> K = klo+16 .. klo+23
        // B fragment: one contiguous 32B global load per lane (pre-swizzled).
        v16h b = *(const v16h*)(Bfrag + (((kt * NT + nt) * 32 + l) << 4));
        acc = __builtin_amdgcn_wmma_f32_16x16x32_f16(
            false, a.v, false, b, (short)0, acc, false, false);
      }
      // D layout: element v holds row m = mt*16 + 8*lhalf + v, col n = lane&15.
      #pragma unroll
      for (int v = 0; v < 8; ++v) {
        int m = mt * 16 + lhalf * 8 + v;
        float tv = fast_tanh(acc[v] + bias);
        mx = (m < 58) ? fmaxf(mx, tv) : mx;   // predicated, no branch
      }
    }
    // Channel n lives in lanes l and l+16: combine halves, lanes 0-15 write.
    mx = fmaxf(mx, __shfl_xor(mx, 16, 32));
    if (lhalf == 0 && n < OUT_C) feats[sid * OUT_C + n] = mx;
  }
}

// ---------------------------------------------------------------------------
// Kernel 2: per-bag attention.  logits[r] = sum_s softmax_s(score[r,:]) * score[r,s]
// (fr never materialized), then log_softmax over r.  One block per bag.
// ---------------------------------------------------------------------------
__global__ __launch_bounds__(256)
void pcnn_attn(const float* __restrict__ feats, const float* __restrict__ r_embed,
               float* __restrict__ out) {
  __shared__ float sf[16 * 232];            // feats, stride 232 vs bank conflicts
  __shared__ float sc[N_REL * 16];          // scores
  __shared__ float slg[N_REL];              // per-relation logits
  __shared__ float snorm;
  const int b = blockIdx.x, tid = threadIdx.x;

  for (int i = tid; i < 16 * OUT_C; i += 256) {
    int s = i / OUT_C, c = i - s * OUT_C;
    sf[s * 232 + c] = feats[(b * 16 + s) * OUT_C + c];
  }
  __syncthreads();

  for (int i = tid; i < N_REL * 16; i += 256) {
    int r = i >> 4, s = i & 15;
    const float* re = r_embed + r * OUT_C;
    const float* fv = sf + s * 232;
    float acc = 0.f;
    #pragma unroll 5
    for (int c = 0; c < OUT_C; ++c) acc = fmaf(re[c], fv[c], acc);
    sc[i] = acc;
  }
  __syncthreads();

  if (tid < N_REL) {
    const float* sr = sc + tid * 16;
    float m = sr[0];
    for (int s = 1; s < 16; ++s) m = fmaxf(m, sr[s]);
    float den = 0.f, num = 0.f;
    for (int s = 0; s < 16; ++s) {
      float e = __expf(sr[s] - m);
      den += e; num += e * sr[s];
    }
    slg[tid] = num / den;
  }
  __syncthreads();

  if (tid == 0) {
    float m = slg[0];
    for (int r = 1; r < N_REL; ++r) m = fmaxf(m, slg[r]);
    float se = 0.f;
    for (int r = 0; r < N_REL; ++r) se += __expf(slg[r] - m);
    snorm = m + __logf(se);
  }
  __syncthreads();

  if (tid < N_REL) out[b * N_REL + tid] = slg[tid] - snorm;
}

// ---------------------------------------------------------------------------
extern "C" void kernel_launch(void* const* d_in, const int* in_sizes, int n_in,
                              void* d_out, int out_size, void* d_ws, size_t ws_size,
                              hipStream_t stream) {
  (void)in_sizes; (void)n_in; (void)out_size; (void)ws_size;
  const int*   tokens     = (const int*)d_in[0];
  const int*   pos1       = (const int*)d_in[1];
  const int*   pos2       = (const int*)d_in[2];
  const int*   sent_lens  = (const int*)d_in[3];
  const float* word_embed = (const float*)d_in[4];
  const float* pos_embed  = (const float*)d_in[5];
  const float* conv_w     = (const float*)d_in[6];
  const float* conv_b     = (const float*)d_in[7];
  const float* r_embed    = (const float*)d_in[8];
  float* out = (float*)d_out;

  _Float16* Bfrag = (_Float16*)d_ws;                       // 92160 B
  float*    feats = (float*)((char*)d_ws + 92160);         // 8192*230*4 B

  pcnn_pack_w<<<(BFRAG_ELEMS + 255) / 256, 256, 0, stream>>>(conv_w, Bfrag);
  pcnn_conv<<<512 * 16, 256, 0, stream>>>(tokens, pos1, pos2, sent_lens,
                                          word_embed, pos_embed, conv_b,
                                          Bfrag, feats);
  pcnn_attn<<<512, 256, 0, stream>>>(feats, r_embed, out);
}
